// ChebEdgeDecoder_26706106646651
// MI455X (gfx1250) — matrix operand order repaired and checked by reading, not compiled, not run on hardware
//
#include <hip/hip_runtime.h>

// ---------------------------------------------------------------------------
// ChebConv GNN decoder for MI455X (gfx1250, wave32, WMMA bf16)
//   h0 = relu(sum_k Tx_k(z) Wc0[k] + z Wl0 + b)     [N,64]  -> [N,128]
//   h1 = relu(...)                                  [N,128] -> [N,128]
//   out = ...                                       [N,128] -> [N,350]
// Linear term folded into Cheb slot 0; each layer = one [N, K*IN]x[K*IN,OUT]
// bf16 WMMA GEMM (32x32 C tile per wave: 2 loads per v_wmma) fed by f32
// sparse propagation with native f32 global atomics.
// ---------------------------------------------------------------------------

typedef __attribute__((ext_vector_type(16))) __bf16 v16bf;
typedef __attribute__((ext_vector_type(8)))  __bf16 v8bf;
typedef __attribute__((ext_vector_type(8)))  float  v8f;

static __device__ __forceinline__ unsigned short f2bf(float f) {
  unsigned u = __float_as_uint(f);
  unsigned r = u + 0x7FFFu + ((u >> 16) & 1u);   // round-to-nearest-even
  return (unsigned short)(r >> 16);
}

// ----------------------------- elementwise ---------------------------------
__global__ void k_zero(float* p, int n) {
  int i = blockIdx.x * blockDim.x + threadIdx.x;
  if (i < n) p[i] = 0.0f;
}

// dst = -src  (safe when dst == src)
__global__ void k_negcopy(float* dst, const float* src, int n) {
  int i = blockIdx.x * blockDim.x + threadIdx.x;
  if (i < n) dst[i] = -src[i];
}

// ----------------------------- edge norm -----------------------------------
__global__ void k_deg(const long long* ei, float* deg, int E) {
  int e = blockIdx.x * blockDim.x + threadIdx.x;
  if (e < E) unsafeAtomicAdd(&deg[(int)ei[e]], 1.0f);
}

__global__ void k_rsqrt(float* d, int n) {
  int i = blockIdx.x * blockDim.x + threadIdx.x;
  if (i < n) d[i] = (d[i] > 0.0f) ? rsqrtf(d[i]) : 0.0f;
}

__global__ void k_edgew(const long long* ei, const float* dis, float* w, int E) {
  int e = blockIdx.x * blockDim.x + threadIdx.x;
  if (e < E) w[e] = -dis[(int)ei[e]] * dis[(int)ei[E + e]];
}

// y[row[e], :] += alpha * w[e] * x[col[e], :]   (float4 per thread)
__global__ void k_scatter(const long long* ei, const float* w,
                          const float* x, float* y, int E, int F4, float alpha) {
  int i = blockIdx.x * blockDim.x + threadIdx.x;
  if (i >= E * F4) return;
  int e = i / F4, f = i % F4;
  int r = (int)ei[e], c = (int)ei[E + e];
  float ww = alpha * w[e];
  const float4 xv = ((const float4*)(x + (size_t)c * F4 * 4))[f];
  float* yp = y + (size_t)r * F4 * 4 + (size_t)f * 4;
  unsafeAtomicAdd(yp + 0, ww * xv.x);
  unsafeAtomicAdd(yp + 1, ww * xv.y);
  unsafeAtomicAdd(yp + 2, ww * xv.z);
  unsafeAtomicAdd(yp + 3, ww * xv.w);
}

// Abf[n, off + j] = bf16(src[n, j]), row stride = `stride` bf16 elements
__global__ void k_cast(const float* src, unsigned short* dst,
                       int NF, int F, int stride, int off) {
  int i = blockIdx.x * blockDim.x + threadIdx.x;
  if (i >= NF) return;
  int n = i / F, j = i % F;
  dst[(size_t)n * stride + off + j] = f2bf(src[i]);
}

// Wt[col, k*IN+i] = bf16(Wc[k,i,col] + (k==0 ? Wl[i,col] : 0)); zero-pad cols
__global__ void k_prepw(const float* Wc, const float* Wl, unsigned short* Wt,
                        int KIN, int IN, int OUT, int OUTP) {
  int i = blockIdx.x * blockDim.x + threadIdx.x;
  if (i >= OUTP * KIN) return;
  int col = i / KIN, r = i % KIN;
  float v = 0.0f;
  if (col < OUT) {
    v = Wc[(size_t)r * OUT + col];
    if (r < IN) v += Wl[(size_t)r * OUT + col];   // k==0 slot
  }
  Wt[(size_t)col * KIN + r] = f2bf(v);
}

__global__ void k_prepb(const float* bc, const float* bl, float* b, int OUT) {
  int i = blockIdx.x * blockDim.x + threadIdx.x;
  if (i < OUT) b[i] = bc[i] + bl[i];
}

// ----------------------------- WMMA GEMM -----------------------------------
// A fragment (16x32 bf16): lanes 0-15 row M=lane hold K 0-7 & 16-23,
// lanes 16-31 hold K 8-15 & 24-31.
static __device__ __forceinline__ v16bf ldfragA(const unsigned short* p) {
  union { v16bf v; v8bf h[2]; } u;
  u.h[0] = *(const v8bf*)(p);
  u.h[1] = *(const v8bf*)(p + 16);
  return u.v;
}
// B fragment (32x16 bf16): lanes 0-15 col N=lane hold K 0-7 & 16-23,
// lanes 16-31 hold K 8-15 & 24-31 (column-major weight panel).
static __device__ __forceinline__ v16bf ldfragB(const unsigned short* p) {
  union { v16bf v; v8bf h[2]; } u;
  u.h[0] = *(const v8bf*)(p);
  u.h[1] = *(const v8bf*)(p + 16);
  return u.v;
}

static __device__ __forceinline__ v8f wmma_bf16(v16bf a, v16bf b, v8f c) {
  return __builtin_amdgcn_wmma_f32_16x16x32_bf16(false, a, false, b,
                                                 (short)0, c, false, false);
}

// C[M,OUT] = A[M,KIN] * Wt[OUTP,KIN]^T + bias.
// One wave computes a 32x32 C tile (2 M-tiles x 2 N-tiles): per 32-wide K
// step, 8 b128 loads feed 4 v_wmma -> 2 loads per matrix op.
__global__ void k_gemm(const unsigned short* __restrict__ A,
                       const unsigned short* __restrict__ Wt,
                       const float* __restrict__ bias,
                       float* __restrict__ C,
                       int M, int KIN, int OUT, int ngroups, int relu) {
  int wid  = (blockIdx.x * blockDim.x + threadIdx.x) >> 5;   // wave-uniform
  int lane = threadIdx.x & 31;
  int mtiles = (M + 31) >> 5;                                // 32-row supertiles
  if (wid >= mtiles * ngroups) return;                       // uniform exit
  int mt = wid / ngroups;
  int ng = wid % ngroups;
  int l15  = lane & 15;
  int half = lane >> 4;

  int arow0 = mt * 32 + l15;
  int arow1 = arow0 + 16;
  if (arow0 >= M) arow0 = M - 1;                             // clamp loads only
  if (arow1 >= M) arow1 = M - 1;
  const unsigned short* ap0 = A  + (size_t)arow0 * KIN + half * 8;
  const unsigned short* ap1 = A  + (size_t)arow1 * KIN + half * 8;
  const unsigned short* bp0 = Wt + (size_t)(ng * 32 + l15) * KIN + half * 8;
  const unsigned short* bp1 = bp0 + (size_t)16 * KIN;

  v8f acc00 = {0.f,0.f,0.f,0.f,0.f,0.f,0.f,0.f};
  v8f acc01 = acc00, acc10 = acc00, acc11 = acc00;
  for (int k = 0; k < KIN; k += 32) {
    v16bf a0 = ldfragA(ap0 + k);
    v16bf a1 = ldfragA(ap1 + k);
    v16bf w0 = ldfragB(bp0 + k);
    v16bf w1 = ldfragB(bp1 + k);
    acc00 = wmma_bf16(a0, w0, acc00);
    acc01 = wmma_bf16(a0, w1, acc01);
    acc10 = wmma_bf16(a1, w0, acc10);
    acc11 = wmma_bf16(a1, w1, acc11);
  }

  // C/D layout: lanes 0-15 N=lane rows M=half*8+r; lanes 16-31 same cols, M+8
  int row0 = mt * 32 + half * 8;       // M-tile 0; M-tile 1 at +16
  int c0 = ng * 32 + l15;
  int c1 = c0 + 16;
  float bb0 = (c0 < OUT) ? bias[c0] : 0.0f;
  float bb1 = (c1 < OUT) ? bias[c1] : 0.0f;
#pragma unroll
  for (int r = 0; r < 8; ++r) {
    int rowA = row0 + r;
    int rowB = rowA + 16;
    if (rowA < M) {
      if (c0 < OUT) {
        float o = acc00[r] + bb0;
        if (relu) o = fmaxf(o, 0.0f);
        C[(size_t)rowA * OUT + c0] = o;
      }
      if (c1 < OUT) {
        float o = acc01[r] + bb1;
        if (relu) o = fmaxf(o, 0.0f);
        C[(size_t)rowA * OUT + c1] = o;
      }
    }
    if (rowB < M) {
      if (c0 < OUT) {
        float o = acc10[r] + bb0;
        if (relu) o = fmaxf(o, 0.0f);
        C[(size_t)rowB * OUT + c0] = o;
      }
      if (c1 < OUT) {
        float o = acc11[r] + bb1;
        if (relu) o = fmaxf(o, 0.0f);
        C[(size_t)rowB * OUT + c1] = o;
      }
    }
  }
}

// ----------------------------- host driver ---------------------------------
extern "C" void kernel_launch(void* const* d_in, const int* in_sizes, int n_in,
                              void* d_out, int out_size, void* d_ws, size_t ws_size,
                              hipStream_t stream) {
  const int K = 5, BOT = 64;
  const float* z  = (const float*)d_in[0];
  const long long* ei = (const long long*)d_in[1];
  const float* Wc0 = (const float*)d_in[2];  const float* bc0 = (const float*)d_in[3];
  const float* Wl0 = (const float*)d_in[4];  const float* bl0 = (const float*)d_in[5];
  const float* Wc1 = (const float*)d_in[6];  const float* bc1 = (const float*)d_in[7];
  const float* Wl1 = (const float*)d_in[8];  const float* bl1 = (const float*)d_in[9];
  const float* Wc2 = (const float*)d_in[10]; const float* bc2 = (const float*)d_in[11];
  const float* Wl2 = (const float*)d_in[12]; const float* bl2 = (const float*)d_in[13];

  const int N   = in_sizes[0] / BOT;      // 50000
  const int E   = in_sizes[1] / 2;        // 800000
  const int HID = in_sizes[3];            // 128
  const int OUT = in_sizes[11];           // 350
  const int OUTPmax = ((OUT + 31) / 32) * 32;

  // workspace carve-up
  char* ws = (char*)d_ws;
  auto carve = [&](size_t bytes) -> void* {
    void* p = (void*)ws;
    ws += (bytes + 255) & ~(size_t)255;
    return p;
  };
  unsigned short* Abf  = (unsigned short*)carve((size_t)N * (K * HID) * 2); // 64 MB
  float* S1   = (float*)carve((size_t)N * HID * 4);
  float* S2   = (float*)carve((size_t)N * HID * 4);
  float* hA   = (float*)carve((size_t)N * HID * 4);
  float* hB   = (float*)carve((size_t)N * HID * 4);
  float* wE   = (float*)carve((size_t)E * 4);
  float* deg  = (float*)carve((size_t)N * 4);
  unsigned short* Wt = (unsigned short*)carve((size_t)(K * HID) * OUTPmax * 2);
  float* bias = (float*)carve((size_t)OUTPmax * 4);

  const int T = 256;
  auto blk = [&](long long n) { return (unsigned)((n + T - 1) / T); };

  // edge normalization: deg -> d^-1/2 -> w = -dis[row]*dis[col]
  k_zero <<<blk(N), T, 0, stream>>>(deg, N);
  k_deg  <<<blk(E), T, 0, stream>>>(ei, deg, E);
  k_rsqrt<<<blk(N), T, 0, stream>>>(deg, N);
  k_edgew<<<blk(E), T, 0, stream>>>(ei, deg, wE, E);

  auto run_layer = [&](const float* x, const float* Wc, const float* bc,
                       const float* Wl, const float* bl,
                       int IN, int OUTL, float* out, int relu) {
    const int KIN = K * IN;
    const int NF  = N * IN;
    const int F4  = IN / 4;
    const long long EW = (long long)E * F4;

    // slot 0: Tx0 = x
    k_cast<<<blk(NF), T, 0, stream>>>(x, Abf, NF, IN, KIN, 0);
    // Tx1 = prop(x)
    k_zero   <<<blk(NF), T, 0, stream>>>(S1, NF);
    k_scatter<<<blk(EW), T, 0, stream>>>(ei, wE, x, S1, E, F4, 1.0f);
    k_cast   <<<blk(NF), T, 0, stream>>>(S1, Abf, NF, IN, KIN, IN);
    // Tx2 = 2 prop(Tx1) - Tx0
    k_negcopy<<<blk(NF), T, 0, stream>>>(S2, x, NF);
    k_scatter<<<blk(EW), T, 0, stream>>>(ei, wE, S1, S2, E, F4, 2.0f);
    k_cast   <<<blk(NF), T, 0, stream>>>(S2, Abf, NF, IN, KIN, 2 * IN);
    // Tx3 = 2 prop(Tx2) - Tx1   (in-place negate of S1, accumulate)
    k_negcopy<<<blk(NF), T, 0, stream>>>(S1, S1, NF);
    k_scatter<<<blk(EW), T, 0, stream>>>(ei, wE, S2, S1, E, F4, 2.0f);
    k_cast   <<<blk(NF), T, 0, stream>>>(S1, Abf, NF, IN, KIN, 3 * IN);
    // Tx4 = 2 prop(Tx3) - Tx2
    k_negcopy<<<blk(NF), T, 0, stream>>>(S2, S2, NF);
    k_scatter<<<blk(EW), T, 0, stream>>>(ei, wE, S1, S2, E, F4, 2.0f);
    k_cast   <<<blk(NF), T, 0, stream>>>(S2, Abf, NF, IN, KIN, 4 * IN);

    // fused weight panel (Wl folded into k=0) + bias, then WMMA GEMM
    const int OUTP = ((OUTL + 31) / 32) * 32;
    k_prepw<<<blk((long long)OUTP * KIN), T, 0, stream>>>(Wc, Wl, Wt, KIN, IN, OUTL, OUTP);
    k_prepb<<<blk(OUTL), T, 0, stream>>>(bc, bl, bias, OUTL);

    const int mtiles = (N + 31) / 32, ngroups = OUTP / 32;
    const long long waves = (long long)mtiles * ngroups;
    k_gemm<<<blk(waves * 32), T, 0, stream>>>(Abf, Wt, bias, out,
                                              N, KIN, OUTL, ngroups, relu);
  };

  run_layer(z,  Wc0, bc0, Wl0, bl0, BOT, HID, hA, 1);
  run_layer(hA, Wc1, bc1, Wl1, bl1, HID, HID, hB, 1);
  run_layer(hB, Wc2, bc2, Wl2, bl2, HID, OUT, (float*)d_out, 0);

  (void)n_in; (void)out_size; (void)ws_size;
}
